// GNNConv_4063039062081
// MI455X (gfx1250) — compile-verified, matrix operand order checked by reading, not run on hardware
//
#include <hip/hip_runtime.h>

// ---------------------------------------------------------------------------
// CDNA5 (gfx1250) GNN edge-conv, bf16 WMMA path, bf16-swizzled LDS tiles.
//   delta = tanh(lrelu(x@Wh1+bh1)@Wh2+bh2)        (head also via WMMA, padded)
//   e     = lrelu([pos[s]-pos[d]+delta[d], x[s]] @ Wf + bf)
//   agg   = segment_max(e, dst)  (empty -> 0)
//   out   = lrelu(agg@Wg1+bg1)@Wg2 + bg2 + x
// ---------------------------------------------------------------------------

typedef __attribute__((ext_vector_type(16))) __bf16 v16bf;
typedef __attribute__((ext_vector_type(8)))  __bf16 v8bf;
typedef __attribute__((ext_vector_type(2)))  __bf16 v2bf;
typedef __attribute__((ext_vector_type(8)))  float  v8f;

static constexpr int kC     = 128;
static constexpr int kNodes = 50000;
static constexpr int kEdges = 800000;
static constexpr float kNeg = 0.01f;
static constexpr unsigned kOrdNInf = 0x007FFFFFu;  // f2ord(-inf) sentinel

__device__ __forceinline__ float lrelu(float v) { return v > 0.0f ? v : kNeg * v; }

__device__ __forceinline__ __bf16 f32_to_bf16(float f) {
  unsigned u = __builtin_bit_cast(unsigned, f);
  u += 0x7FFFu + ((u >> 16) & 1u);               // round-to-nearest-even
  unsigned short s = (unsigned short)(u >> 16);
  return __builtin_bit_cast(__bf16, s);
}

// pack two f32 into a dword of 2 bf16 (RNE)
__device__ __forceinline__ unsigned pack_bf16(float a, float b) {
#if __has_builtin(__builtin_amdgcn_cvt_pk_bf16_f32)
  v2bf p = __builtin_amdgcn_cvt_pk_bf16_f32(a, b);
  return __builtin_bit_cast(unsigned, p);
#else
  unsigned ua = __builtin_bit_cast(unsigned, a);
  unsigned ub = __builtin_bit_cast(unsigned, b);
  ua = (ua + 0x7FFFu + ((ua >> 16) & 1u)) >> 16;
  ub = (ub + 0x7FFFu + ((ub >> 16) & 1u)) & 0xFFFF0000u;
  return ua | ub;
#endif
}

// monotonic order-preserving f32 <-> u32 map (for atomicMax-based float max)
__device__ __forceinline__ unsigned f2ord(float f) {
  unsigned u = __builtin_bit_cast(unsigned, f);
  return (u & 0x80000000u) ? ~u : (u | 0x80000000u);
}
__device__ __forceinline__ float ord2f(unsigned u) {
  unsigned b = (u & 0x80000000u) ? (u & 0x7FFFFFFFu) : ~u;
  return __builtin_bit_cast(float, b);
}
__device__ __forceinline__ float unmap_agg(unsigned u) {
  return (u == kOrdNInf) ? 0.0f : ord2f(u);
}

// Per-lane K index inside a 32-wide K tile (16-bit A/B WMMA VGPR layout):
// lanes 0-15 hold K {0..7,16..23}, lanes 16-31 hold K {8..15,24..31}.
__device__ __forceinline__ int kmap(int h, int laneHi) {
  int v = h >> 1, p = h & 1;
  int base = (v < 4) ? 0 : 16;
  int vv   = (v < 4) ? v : (v - 4);
  return base + laneHi * 8 + vv * 2 + p;
}

// ---------------------------------------------------------------------------
// LDS tile: 16 rows x 128 bf16, each row as 16 chunks of 8 bf16 (16B),
// chunk index XOR-swizzled by row -> ds_*_b128 hits every bank exactly twice.
// ---------------------------------------------------------------------------
__device__ __forceinline__ int tile_off(int row, int chunk) {
  return row * kC + ((chunk ^ row) & 15) * 8;     // in bf16 elements
}

// A fragment for K-tile kt: two contiguous 8-bf16 runs (chunks ca, ca+2).
__device__ __forceinline__ v16bf load_a(const __bf16* tile, int kt, int lane) {
  const int row = lane & 15, hi = lane >> 4;
  const int ca = kt * 4 + hi;
  v8bf lo = *(const v8bf*)(tile + tile_off(row, ca));
  v8bf hb = *(const v8bf*)(tile + tile_off(row, ca + 2));
  return __builtin_shufflevector(lo, hb, 0, 1, 2, 3, 4, 5, 6, 7,
                                 8, 9, 10, 11, 12, 13, 14, 15);
}

__device__ __forceinline__ void store_h_bf16(__bf16* tile, int row, int col,
                                             float v) {
  tile[tile_off(row, col >> 3) + (col & 7)] = f32_to_bf16(v);
}

// [16 x 128] (LDS, bf16 swizzled) @ [128 x 128] (swizzled bf16 B) -> acc[8].
// All 8 B fragments of a K-step are loaded up front (one deep load clause),
// then the 8 WMMAs issue back-to-back behind a single wait.
__device__ __forceinline__ void gemm_tile(const __bf16* __restrict__ tile,
                                          const __bf16* __restrict__ Wb,
                                          v8f acc[8], int lane) {
#pragma unroll
  for (int kt = 0; kt < 4; ++kt) {
    v16bf a = load_a(tile, kt, lane);
    v16bf b[8];
#pragma unroll
    for (int nt = 0; nt < 8; ++nt)
      b[nt] = *(const v16bf*)(Wb + ((kt * 8 + nt) * 32 + lane) * 16);
#pragma unroll
    for (int nt = 0; nt < 8; ++nt)
      acc[nt] = __builtin_amdgcn_wmma_f32_16x16x32_bf16(
          false, a, false, b[nt], (short)0, acc[nt], false, false);
  }
}

// --------------------------- weight preparation ----------------------------
// sections 0..3: 128x128 weights (16384 bf16 each); section 4: Wh2 padded to
// 128x16 (2048 bf16, cols 3..15 zero). 67584 threads total.
__global__ void __launch_bounds__(256) prep_weights(
    const float* __restrict__ Wh1, const float* __restrict__ Wf,
    const float* __restrict__ Wg1, const float* __restrict__ Wg2,
    const float* __restrict__ Wh2, __bf16* __restrict__ ws) {
  int t = blockIdx.x * 256 + threadIdx.x;
  if (t >= 65536) {                                // Wh2 padded tile
    int idx = t - 65536;                           // 0..2047
    int kt = idx >> 9, within = idx & 511;
    int lane = within >> 4, h = within & 15;
    int laneHi = lane >> 4, col = lane & 15;
    int k = kt * 32 + kmap(h, laneHi);
    ws[65536 + idx] = f32_to_bf16(col < 3 ? Wh2[k * 3 + col] : 0.0f);
    return;
  }
  int wsel = t >> 14;
  int idx  = t & 16383;
  int tile = idx >> 9, within = idx & 511;
  int lane = within >> 4, h = within & 15;
  int kt = tile >> 3, nt = tile & 7;
  int laneHi = lane >> 4;
  int col = nt * 16 + (lane & 15);
  int k   = kt * 32 + kmap(h, laneHi);
  const float* W;
  int krow = k;
  if (wsel == 0)      { W = Wh1; }
  else if (wsel == 1) { W = Wf; krow = k + 3; }   // skip 3 'rel' rows
  else if (wsel == 2) { W = Wg1; }
  else                { W = Wg2; }
  ws[wsel * 16384 + idx] = f32_to_bf16(W[krow * kC + col]);
}

__global__ void __launch_bounds__(256) init_agg(unsigned* __restrict__ aggU) {
  int i = blockIdx.x * 256 + threadIdx.x;
  if (i < kNodes * kC) aggU[i] = kOrdNInf;
}

// ------------------------------ node: delta --------------------------------
__global__ void __launch_bounds__(128) node_delta_kernel(
    const float* __restrict__ x, const __bf16* __restrict__ Wh1b,
    const float* __restrict__ bh1, const __bf16* __restrict__ Wh2b,
    const float* __restrict__ bh2, float* __restrict__ delta) {
  __shared__ __align__(16) __bf16 lds[4][16 * kC];
  const int lane = threadIdx.x & 31, wave = threadIdx.x >> 5;
  const int laneHi = lane >> 4, c = lane & 15;
  const int nTiles = kNodes / 16;
  int tile = blockIdx.x * 4 + wave;
  bool active = tile < nTiles;
  if (!active) tile = nTiles - 1;
  const int base = tile * 16;
  __bf16* tileLds = lds[wave];

  // stage x tile as bf16: pass p covers rows 2p, 2p+1; lane owns one 16B chunk
#pragma unroll
  for (int p = 0; p < 8; ++p) {
    int r = p * 2 + laneHi;
    const float* src = x + (base + r) * kC + c * 8;
    float4 g0 = *(const float4*)(src);
    float4 g1 = *(const float4*)(src + 4);
    uint4 w = {pack_bf16(g0.x, g0.y), pack_bf16(g0.z, g0.w),
               pack_bf16(g1.x, g1.y), pack_bf16(g1.z, g1.w)};
    *(uint4*)(tileLds + tile_off(r, c)) = w;
  }
  __syncthreads();

  v8f acc[8];
#pragma unroll
  for (int i = 0; i < 8; ++i) acc[i] = (v8f)(0.0f);
  gemm_tile(tileLds, Wh1b, acc, lane);
  __syncthreads();

#pragma unroll
  for (int nt = 0; nt < 8; ++nt) {
    int col = nt * 16 + c;
    float b = bh1[col];
#pragma unroll
    for (int v = 0; v < 8; ++v)
      store_h_bf16(tileLds, v + laneHi * 8, col, lrelu(acc[nt][v] + b));
  }
  __syncthreads();

  // head h@Wh2 (128->16, cols 3..15 zero-padded) via WMMA, then tanh
  v8f accD = (v8f)(0.0f);
#pragma unroll
  for (int kt = 0; kt < 4; ++kt) {
    v16bf a = load_a(tileLds, kt, lane);
    v16bf b = *(const v16bf*)(Wh2b + (kt * 32 + lane) * 16);
    accD = __builtin_amdgcn_wmma_f32_16x16x32_bf16(
        false, a, false, b, (short)0, accD, false, false);
  }
  if (active && c < 3) {
    float bb = bh2[c];
#pragma unroll
    for (int v = 0; v < 8; ++v) {
      int row = v + laneHi * 8;
      delta[(base + row) * 3 + c] = tanhf(accD[v] + bb);
    }
  }
}

// ------------------------------ edge message -------------------------------
__global__ void __launch_bounds__(128) edge_kernel(
    const float* __restrict__ x, const float* __restrict__ pos,
    const int* __restrict__ ei, const float* __restrict__ delta,
    const __bf16* __restrict__ Wfxb, const float* __restrict__ Wf,
    const float* __restrict__ bf, unsigned* __restrict__ aggU) {
  __shared__ __align__(16) __bf16 lds[4][16 * kC];
  __shared__ int   srcS[4][16], dstS[4][16];
  __shared__ float relS[4][16][3];
  const int lane = threadIdx.x & 31, wave = threadIdx.x >> 5;
  const int laneHi = lane >> 4, c = lane & 15;
  const int tile = blockIdx.x * 4 + wave;      // 50000 tiles, exact cover
  const int base = tile * 16;

  if (lane < 16) {
    int e = base + lane;
    int s = ei[e];
    int d = ei[kEdges + e];
    srcS[wave][lane] = s;
    dstS[wave][lane] = d;
#pragma unroll
    for (int j = 0; j < 3; ++j)
      relS[wave][lane][j] = pos[s * 3 + j] - pos[d * 3 + j] + delta[d * 3 + j];
  }
  __syncthreads();

  __bf16* tileLds = lds[wave];
#pragma unroll
  for (int p = 0; p < 8; ++p) {                  // gather x[src] rows (L2 hits)
    int r = p * 2 + laneHi;
    const float* src = x + srcS[wave][r] * kC + c * 8;
    float4 g0 = *(const float4*)(src);
    float4 g1 = *(const float4*)(src + 4);
    uint4 w = {pack_bf16(g0.x, g0.y), pack_bf16(g0.z, g0.w),
               pack_bf16(g1.x, g1.y), pack_bf16(g1.z, g1.w)};
    *(uint4*)(tileLds + tile_off(r, c)) = w;
  }
  __syncthreads();

  v8f acc[8];
#pragma unroll
  for (int i = 0; i < 8; ++i) acc[i] = (v8f)(0.0f);
  gemm_tile(tileLds, Wfxb, acc, lane);

#pragma unroll
  for (int nt = 0; nt < 8; ++nt) {
    int col = nt * 16 + c;
    float w0 = Wf[col], w1 = Wf[kC + col], w2 = Wf[2 * kC + col], bb = bf[col];
#pragma unroll
    for (int v = 0; v < 8; ++v) {
      int r = v + laneHi * 8;
      float val = acc[nt][v] + relS[wave][r][0] * w0 + relS[wave][r][1] * w1 +
                  relS[wave][r][2] * w2 + bb;
      val = lrelu(val);
      atomicMax(&aggU[(unsigned)dstS[wave][r] * kC + col], f2ord(val));
    }
  }
}

// ------------------------------ node: output -------------------------------
__global__ void __launch_bounds__(128) final_kernel(
    const unsigned* aggU,                         // aliases `out` (no restrict)
    const float* __restrict__ x, const __bf16* __restrict__ Wg1b,
    const float* __restrict__ bg1, const __bf16* __restrict__ Wg2b,
    const float* __restrict__ bg2, float* out) {
  __shared__ __align__(16) __bf16 lds[4][16 * kC];
  const int lane = threadIdx.x & 31, wave = threadIdx.x >> 5;
  const int laneHi = lane >> 4, c = lane & 15;
  const int nTiles = kNodes / 16;
  int tile = blockIdx.x * 4 + wave;
  bool active = tile < nTiles;
  if (!active) tile = nTiles - 1;
  const int base = tile * 16;
  __bf16* tileLds = lds[wave];

  // unmap aggregate (ordered-u32 -> f32, sentinel -> 0) and stage as bf16
#pragma unroll
  for (int p = 0; p < 8; ++p) {
    int r = p * 2 + laneHi;
    const unsigned* src = aggU + (base + r) * kC + c * 8;
    uint4 u0 = *(const uint4*)(src);
    uint4 u1 = *(const uint4*)(src + 4);
    uint4 w = {pack_bf16(unmap_agg(u0.x), unmap_agg(u0.y)),
               pack_bf16(unmap_agg(u0.z), unmap_agg(u0.w)),
               pack_bf16(unmap_agg(u1.x), unmap_agg(u1.y)),
               pack_bf16(unmap_agg(u1.z), unmap_agg(u1.w))};
    *(uint4*)(tileLds + tile_off(r, c)) = w;
  }
  __syncthreads();

  v8f acc[8];
#pragma unroll
  for (int i = 0; i < 8; ++i) acc[i] = (v8f)(0.0f);
  gemm_tile(tileLds, Wg1b, acc, lane);
  __syncthreads();

#pragma unroll
  for (int nt = 0; nt < 8; ++nt) {
    int col = nt * 16 + c;
    float b = bg1[col];
#pragma unroll
    for (int v = 0; v < 8; ++v)
      store_h_bf16(tileLds, v + laneHi * 8, col, lrelu(acc[nt][v] + b));
  }
  __syncthreads();

  v8f acc2[8];
#pragma unroll
  for (int i = 0; i < 8; ++i) acc2[i] = (v8f)(0.0f);
  gemm_tile(tileLds, Wg2b, acc2, lane);

#pragma unroll
  for (int nt = 0; nt < 8; ++nt) {
    int col = nt * 16 + c;
    float b = bg2[col];
#pragma unroll
    for (int v = 0; v < 8; ++v) {
      int row = base + v + laneHi * 8;
      float val = acc2[nt][v] + b + x[row * kC + col];
      if (active) out[row * kC + col] = val;
    }
  }
}

// ------------------------------- launcher ----------------------------------
extern "C" void kernel_launch(void* const* d_in, const int* in_sizes, int n_in,
                              void* d_out, int out_size, void* d_ws, size_t ws_size,
                              hipStream_t stream) {
  const float* x    = (const float*)d_in[0];
  const float* pos  = (const float*)d_in[1];
  const int*   ei   = (const int*)d_in[2];
  const float* Wh1  = (const float*)d_in[3];
  const float* bh1  = (const float*)d_in[4];
  const float* Wh2  = (const float*)d_in[5];
  const float* bh2  = (const float*)d_in[6];
  const float* Wf   = (const float*)d_in[7];
  const float* bf_  = (const float*)d_in[8];
  const float* Wg1  = (const float*)d_in[9];
  const float* bg1  = (const float*)d_in[10];
  const float* Wg2  = (const float*)d_in[11];
  const float* bg2  = (const float*)d_in[12];

  __bf16* wb = (__bf16*)d_ws;                    // swizzled bf16 weights
  const __bf16* Wh1b = wb;                       // 16384 bf16
  const __bf16* Wfxb = wb + 16384;
  const __bf16* Wg1b = wb + 32768;
  const __bf16* Wg2b = wb + 49152;
  const __bf16* Wh2b = wb + 65536;               // 2048 bf16 (padded head)
  float* delta = (float*)((char*)d_ws + 143360); // 50000 x 3 f32
  unsigned* aggU = (unsigned*)d_out;

  const int nodeBlocks = (kNodes / 16 + 3) / 4;  // 4 wave-tiles per block

  prep_weights<<<264, 256, 0, stream>>>(Wh1, Wf, Wg1, Wg2, Wh2, wb);
  init_agg<<<(kNodes * kC) / 256, 256, 0, stream>>>(aggU);
  node_delta_kernel<<<nodeBlocks, 128, 0, stream>>>(x, Wh1b, bh1, Wh2b, bh2, delta);
  edge_kernel<<<kEdges / 64, 128, 0, stream>>>(x, pos, ei, delta, Wfxb, Wf, bf_, aggU);
  final_kernel<<<nodeBlocks, 128, 0, stream>>>(aggU, x, Wg1b, bg1, Wg2b, bg2,
                                               (float*)d_out);
}
